// BinarizedLinear_75110388072540
// MI455X (gfx1250) — 8-bit matrix operand order repaired and checked by reading, compiled, not run on hardware
//
#include <hip/hip_runtime.h>


typedef __attribute__((ext_vector_type(8))) int v8i;
typedef __attribute__((ext_vector_type(4))) int v4i;

#define D_ 128
#define O_ 512
#define I_ 512

__device__ __forceinline__ int swz_chunk(int row, int col) {
    return (((col >> 4) ^ (row & 15)) << 4) | (col & 15);
}

__global__ __launch_bounds__(256)
void BinarizedLinear_wmma_kernel(const int* __restrict__ xg,
                                 const int* __restrict__ wgt,
                                 const float* __restrict__ bias,
                                 float* __restrict__ out)
{
    __shared__ signed char xs[64 * 512];
    __shared__ signed char ws[64 * 512];

    const int tid   = threadIdx.x;
    const int d     = blockIdx.y;
    const int otile = blockIdx.x * 64;

    const int col  = (tid * 4) & 511;
    const int row0 = (tid * 4) >> 9;
    const int j    = col & 63;
    const int pcol = (col & ~63) + ((j >> 4) & 1) * 32 + ((j >> 5) << 4) + (j & 15);

    const int* xsrc = xg + (size_t)row0 * (D_ * I_) + (size_t)d * I_ + col;
#pragma unroll 8
    for (int k = 0; k < 32; ++k) {
        const int row = row0 + 2 * k;
        const v4i v = *(const v4i*)(xsrc + (size_t)(2 * k) * (D_ * I_));
        const unsigned int comb = (unsigned)v.x | ((unsigned)v.y << 8) |
                                  ((unsigned)v.z << 16) | ((unsigned)v.w << 24);
        const unsigned int pk = ~(comb * 0xFEu);
        *(unsigned int*)(xs + row * 512 + swz_chunk(row, pcol)) = pk;
    }
    const int* wsrc = wgt + (size_t)d * (O_ * I_) + (size_t)(otile + row0) * I_ + col;
#pragma unroll 8
    for (int k = 0; k < 32; ++k) {
        const int row = row0 + 2 * k;
        const v4i v = *(const v4i*)(wsrc + (size_t)(2 * k) * I_);
        const unsigned int comb = (unsigned)v.x | ((unsigned)v.y << 8) |
                                  ((unsigned)v.z << 16) | ((unsigned)v.w << 24);
        const unsigned int pk = ~(comb * 0xFEu);
        *(unsigned int*)(ws + row * 512 + swz_chunk(row, col)) = pk;
    }
    __syncthreads();

    const int wave = tid >> 5;
    const int lane = tid & 31;
    const int half = lane >> 4;
    const int l16  = lane & 15;
    const int m0   = (wave >> 2) * 32;
    const int n0   = (wave & 3) * 16;

    v8i acc0 = {};
    v8i acc1 = {};

    const int ar0 = m0 + l16;
    const int ar1 = ar0 + 16;
    const int br  = n0 + l16;

#pragma unroll
    for (int ks = 0; ks < 8; ++ks) {
        const int kb = ks * 64;
        const int ac = (kb >> 4) + 2 * half;
        const v4i a0lo = *(const v4i*)(xs + ar0 * 512 + (((ac + 0) ^ l16) << 4));
        const v4i a0hi = *(const v4i*)(xs + ar0 * 512 + (((ac + 1) ^ l16) << 4));
        const v4i a1lo = *(const v4i*)(xs + ar1 * 512 + (((ac + 0) ^ l16) << 4));
        const v4i a1hi = *(const v4i*)(xs + ar1 * 512 + (((ac + 1) ^ l16) << 4));
        const v8i a0 = __builtin_shufflevector(a0lo, a0hi, 0, 1, 2, 3, 4, 5, 6, 7);
        const v8i a1 = __builtin_shufflevector(a1lo, a1hi, 0, 1, 2, 3, 4, 5, 6, 7);

        const v4i b0 = *(const v4i*)(ws + br * 512 + swz_chunk(br, kb + 16 * half));
        const v4i b1 = *(const v4i*)(ws + br * 512 + swz_chunk(br, kb + 32 + 16 * half));
        const v8i b  = __builtin_shufflevector(b0, b1, 0, 1, 2, 3, 4, 5, 6, 7);

        acc0 = __builtin_amdgcn_wmma_i32_16x16x64_iu8(true, a0, true, b, acc0, false, false);
        acc1 = __builtin_amdgcn_wmma_i32_16x16x64_iu8(true, a1, true, b, acc1, false, false);
    }

    const int o     = otile + n0 + l16;
    const float thr = 2.0f * bias[d * O_ + o] - (float)I_;
    float* orow     = out + (size_t)d * O_ + o;
#pragma unroll
    for (int r = 0; r < 8; ++r) {
        const int b0 = m0 + r + 8 * half;
        const int b1 = b0 + 16;
        orow[(size_t)b0 * (D_ * O_)] = ((float)acc0[r] > thr) ? 1.0f : 0.0f;
        orow[(size_t)b1 * (D_ * O_)] = ((float)acc1[r] > thr) ? 1.0f : 0.0f;
    }
}

extern "C" void kernel_launch(void* const* d_in, const int* in_sizes, int n_in,
                              void* d_out, int out_size, void* d_ws, size_t ws_size,
                              hipStream_t stream) {
    (void)in_sizes; (void)n_in; (void)out_size; (void)d_ws; (void)ws_size;
    const int*   xg   = (const int*)d_in[0];
    const int*   wgt  = (const int*)d_in[1];
    const float* bias = (const float*)d_in[2];
    float*       out  = (float*)d_out;

    dim3 grid(O_ / 64, D_);
    dim3 block(256);
    BinarizedLinear_wmma_kernel<<<grid, block, 0, stream>>>(xg, wgt, bias, out);
}
